// GAT_10L_uw_54485955117457
// MI455X (gfx1250) — matrix operand order repaired and checked by reading, not compiled, not run on hardware
//
#include <hip/hip_runtime.h>
#include <hip/hip_bf16.h>
#include <stdint.h>

// ---------------- problem constants (match reference) ----------------
#define N_NODES   50000
#define N_EDGES   800000
#define EE        (N_EDGES + N_NODES)   // edges + self loops = 850000
#define CDIM      32
#define NEG_SLOPE 0.2f
#define BN_EPS    1e-5f
#define NBLK_STATS 200                  // blocks for deterministic BN reduction
#define MT        5                     // M-tiles per block in GEMM (5*16=80 rows)

typedef __attribute__((ext_vector_type(16))) _Float16 v16h;
typedef __attribute__((ext_vector_type(8)))  _Float16 v8h;
typedef __attribute__((ext_vector_type(8)))  float    v8f;

// ---------------- small helpers ----------------
__device__ __forceinline__ int edge_src(const int* eidx, int e) {
  return (e < N_EDGES) ? eidx[e] : (e - N_EDGES);
}
__device__ __forceinline__ int edge_dst(const int* eidx, int e) {
  return (e < N_EDGES) ? eidx[N_EDGES + e] : (e - N_EDGES);
}

__global__ void k_fill_i32(int* p, int v, int n) {
  int i = blockIdx.x * blockDim.x + threadIdx.x;
  if (i < n) p[i] = v;
}
__global__ void k_copy_i32(const int* s, int* d, int n) {
  int i = blockIdx.x * blockDim.x + threadIdx.x;
  if (i < n) d[i] = s[i];
}
__global__ void k_f32_to_f16(const float* __restrict__ s, _Float16* __restrict__ d, int n) {
  int i = blockIdx.x * blockDim.x + threadIdx.x;
  if (i < n) d[i] = (_Float16)s[i];
}

// ---------------- CSR build (deterministic after per-node sort) ----------------
__global__ void k_count(const int* __restrict__ eidx, int* __restrict__ counts) {
  int e = blockIdx.x * blockDim.x + threadIdx.x;
  if (e >= EE) return;
  atomicAdd(&counts[edge_dst(eidx, e)], 1);
}
__global__ void k_scan(const int* __restrict__ counts, int* __restrict__ rowptr) {
  if (blockIdx.x == 0 && threadIdx.x == 0) {
    int acc = 0;
    for (int i = 0; i < N_NODES; ++i) { rowptr[i] = acc; acc += counts[i]; }
    rowptr[N_NODES] = acc;
  }
}
__global__ void k_fill_perm(const int* __restrict__ eidx, int* __restrict__ cursor,
                            int* __restrict__ perm) {
  int e = blockIdx.x * blockDim.x + threadIdx.x;
  if (e >= EE) return;
  int pos = atomicAdd(&cursor[edge_dst(eidx, e)], 1);
  perm[pos] = e;
}
// sort each node's edge segment by edge id -> CSR independent of atomic race order
__global__ void k_sort_seg(const int* __restrict__ rowptr, int* __restrict__ perm) {
  int n = blockIdx.x * blockDim.x + threadIdx.x;
  if (n >= N_NODES) return;
  int b = rowptr[n], e2 = rowptr[n + 1];
  for (int i = b + 1; i < e2; ++i) {
    int v = perm[i], j = i - 1;
    while (j >= b && perm[j] > v) { perm[j + 1] = perm[j]; --j; }
    perm[j + 1] = v;
  }
}

// ---------------- B-matrix fragment pre-pack ----------------
// Packs f32 weights W[K,Nout] into WMMA B-fragment order, f16:
//   Bp[((ntile*(K/32) + kblk)*32 + lane)*16 + j] = W[kblk*32 + 16*(lane>>4) + j][ntile*16 + (lane&15)]
// One thread per (ntile, kblk, lane): writes one contiguous 16-half (32B) record.
__global__ void k_pack_b(const float* __restrict__ W, _Float16* __restrict__ Bp,
                         int K, int Nout) {
  int gid = blockIdx.x * blockDim.x + threadIdx.x;
  int nkb = K >> 5;
  int total = (Nout >> 4) * nkb * 32;
  if (gid >= total) return;
  int lane  = gid & 31;
  int kblk  = (gid >> 5) % nkb;
  int ntile = (gid >> 5) / nkb;
  int col   = ntile * 16 + (lane & 15);
  int krow  = kblk * 32 + 16 * (lane >> 4);
  _Float16* dst = Bp + (size_t)gid * 16;
  #pragma unroll
  for (int j = 0; j < 16; ++j)
    dst[j] = (_Float16)W[(size_t)(krow + j) * Nout + col];
}

// ---------------- WMMA GEMM ----------------
// C[M,Nout] = A[M,K](f16, row-major) * B(pre-packed fragments), f32 out.
// block = 256 (8 waves). Block computes an 80-row (MT*16) strip of C.
// Wave w owns output column tile nt = w*16. A strip staged in LDS, shared by all waves.
__global__ void k_gemm_wmma(const _Float16* __restrict__ A, const _Float16* __restrict__ Bp,
                            float* __restrict__ C, int K, int Nout) {
  __shared__ _Float16 As[MT * 16 * 256];   // up to K=256: 40 KB
  int tid  = threadIdx.x;
  int wave = tid >> 5;
  int lane = tid & 31;
  int half = lane >> 4;
  int lr   = lane & 15;
  int m0   = blockIdx.x * (MT * 16);
  int nkb  = K >> 5;

  // cooperative, coalesced A strip load: 80 rows x K halves (8-half chunks)
  int chunksPerRow = K >> 3;
  int nChunks = MT * 16 * chunksPerRow;
  for (int idx = tid; idx < nChunks; idx += 256) {
    int row = idx / chunksPerRow;
    int ch  = idx - row * chunksPerRow;
    *(v8h*)&As[row * K + ch * 8] =
        *(const v8h*)(A + (size_t)(m0 + row) * K + ch * 8);
  }
  __syncthreads();

  int nt = wave * 16;
  if (nt < Nout) {                         // wave-uniform; EXEC all-1s inside
    v8f acc[MT];
    #pragma unroll
    for (int t = 0; t < MT; ++t) acc[t] = (v8f){};

    for (int kb = 0; kb < nkb; ++kb) {
      int k0 = kb << 5;
      // B fragment: one contiguous 32B record per lane
      v16h b = *(const v16h*)(Bp + ((size_t)((nt >> 4) * nkb + kb) * 32 + lane) * 16);
      #pragma unroll
      for (int t = 0; t < MT; ++t) {
        // A fragment = two contiguous 16B runs from LDS
        const _Float16* arow = &As[(t * 16 + lr) * K + k0];
        v8h lo = *(const v8h*)(arow + 16 * half);
        v8h hi = *(const v8h*)(arow + 16 * half + 8);
        v16h a;
        #pragma unroll
        for (int j = 0; j < 8; ++j) { a[j] = lo[j]; a[8 + j] = hi[j]; }
        acc[t] = __builtin_amdgcn_wmma_f32_16x16x32_f16(false, a, false, b,
                                                        (short)0, acc[t], false, false);
      }
    }
    // C/D layout: VGPR r -> M = r + 8*half, N = lr
    #pragma unroll
    for (int t = 0; t < MT; ++t)
      #pragma unroll
      for (int r = 0; r < 8; ++r)
        C[(size_t)(m0 + t * 16 + r + 8 * half) * Nout + nt + lr] = acc[t][r];
  }
}

// ---------------- per-edge attention logits ----------------
// logit[e*h+hh] = sum_c att[hh,c] * leaky(XL[src,hh*32+c] + XR[dst,hh*32+c])
__global__ void k_edge_logit(const int* __restrict__ eidx, const float* __restrict__ XL,
                             const float* __restrict__ XR, const float* __restrict__ att,
                             float* __restrict__ logit, int hshift, int dout) {
  int gid = blockIdx.x * blockDim.x + threadIdx.x;
  int h = 1 << hshift;
  if (gid >= EE * h) return;
  int e = gid >> hshift, hh = gid & (h - 1);
  int s = edge_src(eidx, e), d = edge_dst(eidx, e);
  const float4* xl = (const float4*)(XL + (size_t)s * dout + hh * CDIM);
  const float4* xr = (const float4*)(XR + (size_t)d * dout + hh * CDIM);
  const float4* a4 = (const float4*)(att + hh * CDIM);
  float acc = 0.0f;
  #pragma unroll
  for (int q = 0; q < 8; ++q) {
    float4 l = xl[q], r = xr[q], a = a4[q];
    float v;
    v = l.x + r.x; v = (v >= 0.f) ? v : NEG_SLOPE * v; acc += a.x * v;
    v = l.y + r.y; v = (v >= 0.f) ? v : NEG_SLOPE * v; acc += a.y * v;
    v = l.z + r.z; v = (v >= 0.f) ? v : NEG_SLOPE * v; acc += a.z * v;
    v = l.w + r.w; v = (v >= 0.f) ? v : NEG_SLOPE * v; acc += a.w * v;
  }
  logit[(size_t)e * h + hh] = acc;
}

// ---------------- per-(node,head) softmax + weighted aggregation (no atomics) ----
__global__ void k_node_agg(const int* __restrict__ eidx, const int* __restrict__ rowptr,
                           const int* __restrict__ perm, const float* __restrict__ logit,
                           const float* __restrict__ XL, float* __restrict__ agg,
                           int hshift, int dout) {
  int gid = blockIdx.x * blockDim.x + threadIdx.x;
  int h = 1 << hshift;
  if (gid >= N_NODES * h) return;
  int n = gid >> hshift, hh = gid & (h - 1);
  int b = rowptr[n], en = rowptr[n + 1];
  float m = -3.4e38f;
  for (int j = b; j < en; ++j)
    m = fmaxf(m, logit[(size_t)perm[j] * h + hh]);
  float den = 0.0f;
  float4 acc[8];
  #pragma unroll
  for (int q = 0; q < 8; ++q) acc[q] = make_float4(0.f, 0.f, 0.f, 0.f);
  for (int j = b; j < en; ++j) {
    int e = perm[j];
    float w = expf(logit[(size_t)e * h + hh] - m);
    den += w;
    int s = edge_src(eidx, e);
    const float4* xl = (const float4*)(XL + (size_t)s * dout + hh * CDIM);
    #pragma unroll
    for (int q = 0; q < 8; ++q) {
      float4 v = xl[q];
      acc[q].x += w * v.x; acc[q].y += w * v.y;
      acc[q].z += w * v.z; acc[q].w += w * v.w;
    }
  }
  float inv = 1.0f / den;   // every node has a self-loop -> den > 0
  float4* out = (float4*)(agg + (size_t)n * dout + hh * CDIM);
  #pragma unroll
  for (int q = 0; q < 8; ++q)
    out[q] = make_float4(acc[q].x * inv, acc[q].y * inv, acc[q].z * inv, acc[q].w * inv);
}

// ---------------- bias + relu (in place) + deterministic BN partials ----------------
__global__ void k_bias_relu_stats(float* __restrict__ y, const float* __restrict__ bias,
                                  float* __restrict__ psum, float* __restrict__ psq,
                                  int dout) {
  const int ROWS = (N_NODES + NBLK_STATS - 1) / NBLK_STATS;   // 250
  int r0 = blockIdx.x * ROWS;
  int r1 = min(r0 + ROWS, N_NODES);
  int tpc = blockDim.x / dout;                                // 2 (dout=128) or 8 (dout=32)
  int c = threadIdx.x % dout;
  int k = threadIdx.x / dout;
  float s = 0.f, q = 0.f;
  float bc = bias[c];
  for (int r = r0 + k; r < r1; r += tpc) {
    float v = y[(size_t)r * dout + c] + bc;
    v = fmaxf(v, 0.0f);
    y[(size_t)r * dout + c] = v;
    s += v; q += v * v;
  }
  __shared__ float sm[512];
  sm[threadIdx.x] = s; sm[256 + threadIdx.x] = q;
  __syncthreads();
  if (k == 0) {
    for (int kk = 1; kk < tpc; ++kk) {
      s += sm[c + kk * dout];
      q += sm[256 + c + kk * dout];
    }
    psum[blockIdx.x * dout + c] = s;
    psq [blockIdx.x * dout + c] = q;
  }
}

__global__ void k_bn_finalize(const float* __restrict__ psum, const float* __restrict__ psq,
                              const float* __restrict__ gamma, const float* __restrict__ beta,
                              float* __restrict__ ss, int dout) {
  int c = threadIdx.x;
  if (c >= dout) return;
  float s = 0.f, q = 0.f;
  for (int b = 0; b < NBLK_STATS; ++b) { s += psum[b * dout + c]; q += psq[b * dout + c]; }
  float inv_n = 1.0f / (float)N_NODES;
  float mu = s * inv_n;
  float var = q * inv_n - mu * mu;
  float scale = gamma[c] * rsqrtf(var + BN_EPS);
  ss[c]       = scale;
  ss[128 + c] = beta[c] - mu * scale;
}

// BN apply, dual-write: f32 (for final head / reference dtype) + f16 (next layer GEMM A)
__global__ void k_bn_apply(const float* __restrict__ y, const float* __restrict__ ss,
                           float* __restrict__ xo, _Float16* __restrict__ xh, int dout) {
  int i = blockIdx.x * blockDim.x + threadIdx.x;
  if (i >= N_NODES * dout) return;
  int c = i % dout;
  float v = y[i] * ss[c] + ss[128 + c];
  xo[i] = v;
  xh[i] = (_Float16)v;
}

// ---------------- final linear head: [N,32] @ [32,2] + b ----------------
__global__ void k_head(const float* __restrict__ x, const float* __restrict__ w,
                       const float* __restrict__ b, float* __restrict__ out) {
  int n = blockIdx.x * blockDim.x + threadIdx.x;
  if (n >= N_NODES) return;
  const float* row = x + (size_t)n * CDIM;
  float s0 = b[0], s1 = b[1];
  #pragma unroll
  for (int c = 0; c < CDIM; ++c) {
    float v = row[c];
    s0 += v * w[c * 2 + 0];
    s1 += v * w[c * 2 + 1];
  }
  out[(size_t)n * 2 + 0] = s0;
  out[(size_t)n * 2 + 1] = s1;
}

// ---------------- host orchestration ----------------
extern "C" void kernel_launch(void* const* d_in, const int* in_sizes, int n_in,
                              void* d_out, int out_size, void* d_ws, size_t ws_size,
                              hipStream_t stream) {
  (void)in_sizes; (void)n_in; (void)out_size; (void)ws_size;
  const float* x0   = (const float*)d_in[0];
  const int*   eidx = (const int*)d_in[1];
  const float* Wl[10], *Wr[10], *att[10], *bias[10], *gam[10], *bet[10];
  for (int i = 0; i < 10; ++i) {
    Wl[i]   = (const float*)d_in[2  + i];
    Wr[i]   = (const float*)d_in[12 + i];
    att[i]  = (const float*)d_in[22 + i];
    bias[i] = (const float*)d_in[32 + i];
    gam[i]  = (const float*)d_in[42 + i];
    bet[i]  = (const float*)d_in[52 + i];
  }
  const float* lin_w = (const float*)d_in[62];
  const float* lin_b = (const float*)d_in[63];

  // ---- carve workspace (aligned 256B) ----
  size_t off = 0;
  auto take = [&](size_t bytes) -> void* {
    void* p = (void*)((uint8_t*)d_ws + off);
    off += (bytes + 255) & ~(size_t)255;
    return p;
  };
  float*    bufX   = (float*)   take((size_t)N_NODES * 128 * 4); // current features (f32)
  _Float16* bufXh  = (_Float16*)take((size_t)N_NODES * 256 * 2); // f16 features (GEMM A)
  float*    bufXL  = (float*)   take((size_t)N_NODES * 128 * 4);
  float*    bufXR  = (float*)   take((size_t)N_NODES * 128 * 4); // also reused as agg / y
  float*    bufP   = (float*)   take((size_t)EE * 4 * 4);        // logits [EE,h]
  int*      rowptr = (int*)     take((size_t)(N_NODES + 1) * 4);
  int*      cursor = (int*)     take((size_t)N_NODES * 4);       // counts, then cursors
  int*      perm   = (int*)     take((size_t)EE * 4);
  float*    psum   = (float*)   take((size_t)NBLK_STATS * 128 * 4);
  float*    psq    = (float*)   take((size_t)NBLK_STATS * 128 * 4);
  float*    ssbuf  = (float*)   take(256 * 4);
  _Float16* wlp    = (_Float16*)take((size_t)256 * 128 * 2);     // packed B fragments
  _Float16* wrp    = (_Float16*)take((size_t)256 * 128 * 2);

  const int B = 256;
  auto nb = [](long n, int b) { return (int)((n + b - 1) / b); };

  // ---- build CSR over dst (once per call; graph is fixed across layers) ----
  k_fill_i32 <<<nb(N_NODES, B), B, 0, stream>>>(cursor, 0, N_NODES);
  k_count    <<<nb(EE, B),      B, 0, stream>>>(eidx, cursor);
  k_scan     <<<1, 1,              0, stream>>>(cursor, rowptr);
  k_copy_i32 <<<nb(N_NODES, B), B, 0, stream>>>(rowptr, cursor, N_NODES);
  k_fill_perm<<<nb(EE, B),      B, 0, stream>>>(eidx, cursor, perm);
  k_sort_seg <<<nb(N_NODES, B), B, 0, stream>>>(rowptr, perm);

  for (int i = 0; i < 10; ++i) {
    int din    = (i == 0) ? 256 : 128;
    int hshift = (i < 9) ? 2 : 0;
    int h      = 1 << hshift;
    int dout   = h * CDIM;   // 128 or 32

    if (i == 0)  // later layers: bn_apply already produced f16 features
      k_f32_to_f16<<<nb((long)N_NODES * din, B), B, 0, stream>>>(x0, bufXh, N_NODES * din);

    // pack weights into B-fragment order (f16)
    int packThreads = (dout >> 4) * (din >> 5) * 32;
    k_pack_b<<<nb(packThreads, B), B, 0, stream>>>(Wl[i], wlp, din, dout);
    k_pack_b<<<nb(packThreads, B), B, 0, stream>>>(Wr[i], wrp, din, dout);

    // XL = Xh @ Wl, XR = Xh @ Wr  (WMMA f16 -> f32, LDS-staged A, 5 M-tiles/block)
    dim3 g(N_NODES / (MT * 16), 1);   // 625 blocks
    k_gemm_wmma<<<g, B, 0, stream>>>(bufXh, wlp, bufXL, din, dout);
    k_gemm_wmma<<<g, B, 0, stream>>>(bufXh, wrp, bufXR, din, dout);

    // per-edge logits
    k_edge_logit<<<nb((long)EE * h, B), B, 0, stream>>>(eidx, bufXL, bufXR, att[i],
                                                        bufP, hshift, dout);
    // per-(node,head) softmax + aggregation -> agg (reuses bufXR; XR no longer needed)
    k_node_agg<<<nb((long)N_NODES * h, B), B, 0, stream>>>(eidx, rowptr, perm, bufP,
                                                           bufXL, bufXR, hshift, dout);
    // bias + relu (in place) + BN partial stats, finalize, apply -> bufX (f32) + bufXh (f16)
    k_bias_relu_stats<<<NBLK_STATS, B, 0, stream>>>(bufXR, bias[i], psum, psq, dout);
    k_bn_finalize<<<1, 128, 0, stream>>>(psum, psq, gam[i], bet[i], ssbuf, dout);
    k_bn_apply<<<nb((long)N_NODES * dout, B), B, 0, stream>>>(bufXR, ssbuf, bufX, bufXh, dout);
  }

  // final linear head -> d_out [N,2] f32
  k_head<<<nb(N_NODES, B), B, 0, stream>>>(bufX, lin_w, lin_b, (float*)d_out);
}
